// RM3ExpertPool_9809705304335
// MI455X (gfx1250) — matrix operand order repaired and checked by reading, not compiled, not run on hardware
//
#include <hip/hip_runtime.h>

#define E_ 8
#define B_ 16384
#define D_ 384
#define R_ 32
#define BM 32            // batch rows per block
#define KT_D 12          // D/32 k-steps

typedef __attribute__((ext_vector_type(16))) __bf16 v16bf;
typedef __attribute__((ext_vector_type(8)))  float  v8f;
typedef __attribute__((ext_vector_type(4)))  unsigned int u32x4;
typedef __attribute__((ext_vector_type(8)))  int i32x8;
typedef __attribute__((ext_vector_type(4)))  int i32x4;

union Frag16 { v16bf v; uint4 q[2]; unsigned int u[8]; };

// ---- LDS layout (bytes) ----
#define XT_OFF     0
#define XT_STRIDE  816       // 768B data + 3 x 16B TDM pads (pad every 256B)
#define XAL_OFF    26128
#define XAL_STRIDE 784       // 768B + 16B tail pad
#define WPA_OFF    51216     // weight panel buffer A (double-buffered)
#define WPB_OFF    81936     // weight panel buffer B
#define WP_STRIDE  80        // 64B k-pairs + 16B pad per n-row
#define P_OFF      112656
#define P_STRIDE   528       // 132 floats
#define NRE_OFF    129552    // 8 experts * 32 rows * 64B
#define ERR_OFF    145936
#define BESTE_OFF  146064
#define BERR_OFF   146192
#define LDS_TOTAL  146432

// ---- workspace layout (bytes): pre-converted bf16 WMMA panels ----
#define WS_APAN    0                          // A_align panels: [e][kk] 384 rows * 80B
#define APAN_BYTES 30720
#define WS_CPAN    2949120                    // Wcat panels: [e][kk] 128 rows * 80B
#define CPAN_BYTES 10240
#define WS_OPAN    3932160                    // W_out panels: [e] 384 rows * 80B
#define OPAN_BYTES 30720
#define WS_XB      4177920                    // x as bf16 row-major [B][384]
#define XB_STRIDE  768
#define WS_TOTAL   16760832

static __device__ __forceinline__ unsigned int pack_bf16(float a, float b) {
    unsigned int ua = __float_as_uint(a), ub = __float_as_uint(b);
    ua = (ua + 0x7FFFu + ((ua >> 16) & 1u)) >> 16;      // RNE
    ub = (ub + 0x7FFFu + ((ub >> 16) & 1u)) >> 16;
    return (ua & 0xFFFFu) | (ub << 16);
}
static __device__ __forceinline__ unsigned short bf16_of(float a) {
    unsigned int ua = __float_as_uint(a);
    return (unsigned short)((ua + 0x7FFFu + ((ua >> 16) & 1u)) >> 16);
}
static __device__ __forceinline__ float bf16_to_f32(unsigned short h) {
    return __uint_as_float(((unsigned int)h) << 16);
}
static __device__ __forceinline__ v8f vzero8() {
    v8f z = {0.f, 0.f, 0.f, 0.f, 0.f, 0.f, 0.f, 0.f};
    return z;
}

// ---- Tensor Data Mover: build D# and issue TENSOR_LOAD_TO_LDS ----
// data_size = 8B units. tile_d1==0 -> 1D contiguous copy. pad_ctl packs
// pad_enable(bit20) | pad_interval(bits24:22) | pad_amount(bits31:25).
static __device__ __forceinline__ void tdm_load(unsigned lds_addr, const void* g,
                                                unsigned tile_d0, unsigned tile_d1,
                                                unsigned stride_d0, unsigned pad_ctl) {
    unsigned long long ga = (unsigned long long)(size_t)g;
    unsigned td1 = tile_d1 ? tile_d1 : 1u;
    u32x4 g0;
    g0.x = 1u;                                               // count=1, user mode
    g0.y = lds_addr;                                         // lds_addr
    g0.z = (unsigned)ga;                                     // global_addr lo
    g0.w = ((unsigned)(ga >> 32) & 0x1FFFFFFu) | 0x80000000u; // global_addr hi | type=2
    i32x8 g1;
    g1[0] = (int)((3u << 16) | pad_ctl);                     // data_size=8B, pad ctl
    g1[1] = (int)((tile_d0 & 0xFFFFu) << 16);                // tensor_dim0 lo16
    g1[2] = (int)(((tile_d0 >> 16) & 0xFFFFu) | ((td1 & 0xFFFFu) << 16)); // td0 hi | td1 lo
    g1[3] = (int)(((td1 >> 16) & 0xFFFFu) | ((tile_d0 & 0xFFFFu) << 16)); // td1 hi | tile_dim0
    g1[4] = (int)(tile_d1 & 0xFFFFu);                        // tile_dim1 (0 => 1D)
    g1[5] = (int)stride_d0;                                  // tensor_dim0_stride lo32
    g1[6] = 0;
    g1[7] = 0;
    i32x4 z4 = {0, 0, 0, 0};
#if defined(__clang_major__) && (__clang_major__ >= 23)
    i32x8 z8 = {0, 0, 0, 0, 0, 0, 0, 0};
    __builtin_amdgcn_tensor_load_to_lds(g0, g1, z4, z4, z8, 0);
#else
    __builtin_amdgcn_tensor_load_to_lds(g0, g1, z4, z4, 0);
#endif
}

// =====================  prep kernel 1: weights -> bf16 panels  =====================
__global__ __launch_bounds__(256)
void rm3_prep_weights(const float* __restrict__ A_align, const float* __restrict__ W_in,
                      const float* __restrict__ W_dt, const float* __restrict__ W_phase,
                      const float* __restrict__ W_out, unsigned char* __restrict__ ws)
{
    int idx = blockIdx.x * 256 + threadIdx.x;
    const int NA = E_ * KT_D * 16 * D_;      // 589824
    const int NC = E_ * KT_D * 128 * 16;     // 196608
    const int NO = E_ * D_ * 16;             // 49152
    if (idx < NA) {
        int f = idx % D_; int t = idx / D_;
        int p = t & 15; t >>= 4;
        int kk = t % KT_D; int e = t / KT_D;
        const float* src = A_align + ((size_t)e * D_ + kk * 32 + 2 * p) * D_ + f;
        *(unsigned int*)(ws + WS_APAN + (size_t)(e * KT_D + kk) * APAN_BYTES
                         + f * WP_STRIDE + p * 4) = pack_bf16(src[0], src[D_]);
    } else if ((idx -= NA) < NC) {
        int p = idx & 15; int t = idx >> 4;
        int n = t & 127; t >>= 7;
        int kk = t % KT_D; int e = t / KT_D;
        const float* wsrc;
        if (n < 64)      wsrc = W_in    + ((size_t)e * 128 + n)       * D_;
        else if (n < 96) wsrc = W_dt    + ((size_t)e * 32 + (n - 64)) * D_;
        else             wsrc = W_phase + ((size_t)e * 32 + (n - 96)) * D_;
        float2 f2 = *(const float2*)(wsrc + kk * 32 + 2 * p);
        *(unsigned int*)(ws + WS_CPAN + (size_t)(e * KT_D + kk) * CPAN_BYTES
                         + n * WP_STRIDE + p * 4) = pack_bf16(f2.x, f2.y);
    } else if ((idx -= NC) < NO) {
        int p = idx & 15; int t = idx >> 4;
        int d = t % D_; int e = t / D_;
        float2 f2 = *(const float2*)(W_out + ((size_t)e * D_ + d) * R_ + 2 * p);
        *(unsigned int*)(ws + WS_OPAN + (size_t)e * OPAN_BYTES
                         + d * WP_STRIDE + p * 4) = pack_bf16(f2.x, f2.y);
    }
}

// =====================  prep kernel 2: x -> bf16 row-major  =====================
__global__ __launch_bounds__(256)
void rm3_prep_x(const float* __restrict__ x, unsigned char* __restrict__ ws)
{
    size_t i = (size_t)blockIdx.x * 256 + threadIdx.x;     // dword index, B*D/2 total
    float2 f = *(const float2*)(x + i * 2);
    *(unsigned int*)(ws + WS_XB + i * 4) = pack_bf16(f.x, f.y);
}

// =====================  main fused kernel  =====================
__global__ __launch_bounds__(256)
void rm3_expert_pool_kernel(const unsigned char* __restrict__ ws,
                            const float* __restrict__ state_re,
                            const float* __restrict__ state_im,
                            const float* __restrict__ log_omega,
                            const float* __restrict__ log_temp,
                            const float* __restrict__ nu,
                            float* __restrict__ out)
{
    extern __shared__ char smem[];
    const int tid  = threadIdx.x;
    const int lane = tid & 31;
    const int wv   = tid >> 5;
    const int rb   = blockIdx.x * BM;
    const int rr   = lane & 15;
    const int hsel = (lane < 16) ? 0 : 1;

#define LDSA(off) ((unsigned)(size_t)(smem + (off)))

    // x tile: one 2D TDM with LDS padding (row 768B, +16B pad each 256B -> stride 816)
    if (wv == 0) {
        tdm_load(LDSA(XT_OFF), ws + WS_XB + (size_t)rb * XB_STRIDE,
                 96, 32, 96, (1u << 20) | (5u << 22) | (3u << 25));
    }
    if (tid < BM) {
        *(float*)(smem + ERR_OFF  + tid * 4) = 0.f;
        *(float*)(smem + BERR_OFF + tid * 4) = 3.0e38f;
        *(int*)  (smem + BESTE_OFF + tid * 4) = 0;
    }

    const int m1 = wv >> 2;          // 16-row tile (0..1)
    const int j1 = (wv & 3) * 6;     // 6 n-tiles of 16 per wave

    // ============================ expert loop ============================
    for (int e = 0; e < E_; ++e) {
        // ---------- GEMM1: x_al = x @ A_align[e]  (TDM double-buffered panels) ----------
        const unsigned char* apan_e = ws + WS_APAN + (size_t)e * KT_D * APAN_BYTES;
        if (wv == 0) tdm_load(LDSA(WPA_OFF), apan_e, 3840, 0, 3840, 0);

        v8f acc1[6];
#pragma unroll
        for (int j = 0; j < 6; ++j) acc1[j] = vzero8();

        for (int kk = 0; kk < KT_D; ++kk) {
            if (wv == 0) {
                if (kk + 1 < KT_D) {
                    tdm_load(LDSA((kk & 1) ? WPA_OFF : WPB_OFF),
                             apan_e + (size_t)(kk + 1) * APAN_BYTES, 3840, 0, 3840, 0);
                    __builtin_amdgcn_s_wait_tensorcnt(1);   // panel kk done, kk+1 in flight
                } else {
                    __builtin_amdgcn_s_wait_tensorcnt(0);
                }
            }
            __syncthreads();
            const unsigned wp = (kk & 1) ? WPB_OFF : WPA_OFF;

            Frag16 a;
            const char* ap = smem + XT_OFF + (m1 * 16 + rr) * XT_STRIDE
                             + kk * 64 + ((kk >> 2) << 4) + hsel * 16;
            a.q[0] = *(const uint4*)ap;
            a.q[1] = *(const uint4*)(ap + 32);
#pragma unroll
            for (int j = 0; j < 6; ++j) {
                Frag16 b;
                const char* bp = smem + wp + ((j1 + j) * 16 + rr) * WP_STRIDE + hsel * 16;
                b.q[0] = *(const uint4*)bp;
                b.q[1] = *(const uint4*)(bp + 32);
                acc1[j] = __builtin_amdgcn_wmma_f32_16x16x32_bf16(
                    false, a.v, false, b.v, (short)0, acc1[j], false, false);
            }
            __syncthreads();
        }

        // ---------- err (||x - x_al||^2) + store x_al as bf16 ----------
        {
            float errp[8];
#pragma unroll
            for (int v = 0; v < 8; ++v) errp[v] = 0.f;
#pragma unroll
            for (int j = 0; j < 6; ++j) {
                int col = (j1 + j) * 16 + rr;
                int cb  = 2 * col + ((col >> 7) << 4);   // padded XT byte offset of col
#pragma unroll
                for (int v = 0; v < 8; ++v) {
                    int row = m1 * 16 + (hsel ? v + 8 : v);
                    float xv = bf16_to_f32(*(const unsigned short*)(smem + XT_OFF + row * XT_STRIDE + cb));
                    float d  = xv - acc1[j][v];
                    errp[v] += d * d;
                    *(unsigned short*)(smem + XAL_OFF + row * XAL_STRIDE + col * 2) = bf16_of(acc1[j][v]);
                }
            }
#pragma unroll
            for (int v = 0; v < 8; ++v) {
                int row = m1 * 16 + (hsel ? v + 8 : v);
                atomicAdd((float*)(smem + ERR_OFF + row * 4), errp[v]);
            }
        }
        __syncthreads();
        if (tid < BM) {
            float er = *(float*)(smem + ERR_OFF + tid * 4);
            *(float*)(smem + ERR_OFF + tid * 4) = 0.f;
            if (er < *(float*)(smem + BERR_OFF + tid * 4)) {   // argmin err == argmax fidelity
                *(float*)(smem + BERR_OFF + tid * 4) = er;
                *(int*)(smem + BESTE_OFF + tid * 4) = e;
            }
        }
        __syncthreads();

        // ---------- GEMM2: P = x_al @ [W_in[0:64]; W_dt; W_phase] ----------
        const unsigned char* cpan_e = ws + WS_CPAN + (size_t)e * KT_D * CPAN_BYTES;
        if (wv == 0) tdm_load(LDSA(WPA_OFF), cpan_e, 1280, 0, 1280, 0);

        const int t0 = wv * 2;
        const int m2 = t0 >> 3;
        v8f acc2[2];
        acc2[0] = vzero8(); acc2[1] = vzero8();

        for (int kk = 0; kk < KT_D; ++kk) {
            if (wv == 0) {
                if (kk + 1 < KT_D) {
                    tdm_load(LDSA((kk & 1) ? WPA_OFF : WPB_OFF),
                             cpan_e + (size_t)(kk + 1) * CPAN_BYTES, 1280, 0, 1280, 0);
                    __builtin_amdgcn_s_wait_tensorcnt(1);
                } else {
                    __builtin_amdgcn_s_wait_tensorcnt(0);
                }
            }
            __syncthreads();
            const unsigned wp = (kk & 1) ? WPB_OFF : WPA_OFF;

            Frag16 a;
            const char* ap = smem + XAL_OFF + (m2 * 16 + rr) * XAL_STRIDE + kk * 64 + hsel * 16;
            a.q[0] = *(const uint4*)ap;
            a.q[1] = *(const uint4*)(ap + 32);
#pragma unroll
            for (int u2 = 0; u2 < 2; ++u2) {
                int n = (t0 + u2) & 7;
                Frag16 b;
                const char* bp = smem + wp + (n * 16 + rr) * WP_STRIDE + hsel * 16;
                b.q[0] = *(const uint4*)bp;
                b.q[1] = *(const uint4*)(bp + 32);
                acc2[u2] = __builtin_amdgcn_wmma_f32_16x16x32_bf16(
                    false, a.v, false, b.v, (short)0, acc2[u2], false, false);
            }
            __syncthreads();
        }
#pragma unroll
        for (int u2 = 0; u2 < 2; ++u2) {
            int t = t0 + u2, mb = (t >> 3) * 16, nb = (t & 7) * 16;
#pragma unroll
            for (int v = 0; v < 8; ++v) {
                int row = mb + (hsel ? v + 8 : v);
                *(float*)(smem + P_OFF + row * P_STRIDE + (nb + rr) * 4) = acc2[u2][v];
            }
        }
        __syncthreads();

        // ---------- elementwise oscillator update -> new_re[e] (bf16 in LDS) ----------
        {
            int row = tid >> 3;
            int r0  = (tid & 7) * 4;
            const char* pr = smem + P_OFF + row * P_STRIDE;
            float g[4], xv[4], dv[4], ph[4];
            *(float4*)g  = *(const float4*)(pr + r0 * 4);
            *(float4*)xv = *(const float4*)(pr + (32 + r0) * 4);
            *(float4*)dv = *(const float4*)(pr + (64 + r0) * 4);
            *(float4*)ph = *(const float4*)(pr + (96 + r0) * 4);
            const size_t sidx = ((size_t)e * B_ + rb + row) * R_ + r0;
            float sre[4], sim[4];
            *(float4*)sre = *(const float4*)(state_re + sidx);
            *(float4*)sim = *(const float4*)(state_im + sidx);
            float nr[4];
#pragma unroll
            for (int i = 0; i < 4; ++i) {
                int r = r0 + i;
                float phi = __expf(log_omega[e * R_ + r]) / (__expf(log_temp[e * R_ + r]) + 1e-6f);
                float dt  = (dv[i] > 20.f) ? dv[i] : log1pf(__expf(dv[i]));   // softplus
                float phs = tanhf(ph[i]) * 3.14159265358979323846f;
                float ang = dt * (phi + phs);
                float dec = __expf(-dt * __expf(nu[e * R_ + r]));
                float s, c;
                __sincosf(ang, &s, &c);
                float sig = 1.f / (1.f + __expf(-g[i]));
                nr[i] = (sre[i] * c - sim[i] * s) * dec + sig * xv[i];
            }
            char* np = smem + NRE_OFF + e * 2048 + row * 64 + r0 * 2;
            *(unsigned int*)np       = pack_bf16(nr[0], nr[1]);
            *(unsigned int*)(np + 4) = pack_bf16(nr[2], nr[3]);
        }
        __syncthreads();
    } // expert loop

    // ---------- winner-masked output GEMM: out = new_re[winner] @ W_out^T ----------
    v8f acc3[6];
#pragma unroll
    for (int j = 0; j < 6; ++j) acc3[j] = vzero8();

    if (wv == 0) tdm_load(LDSA(WPA_OFF), ws + WS_OPAN, 3840, 0, 3840, 0);

    for (int e = 0; e < E_; ++e) {
        if (wv == 0) {
            if (e + 1 < E_) {
                tdm_load(LDSA((e & 1) ? WPA_OFF : WPB_OFF),
                         ws + WS_OPAN + (size_t)(e + 1) * OPAN_BYTES, 3840, 0, 3840, 0);
                __builtin_amdgcn_s_wait_tensorcnt(1);
            } else {
                __builtin_amdgcn_s_wait_tensorcnt(0);
            }
        }
        __syncthreads();
        const unsigned wp = (e & 1) ? WPB_OFF : WPA_OFF;

        Frag16 a;
        int arow = m1 * 16 + rr;
        const char* ap = smem + NRE_OFF + e * 2048 + arow * 64 + hsel * 16;
        a.q[0] = *(const uint4*)ap;
        a.q[1] = *(const uint4*)(ap + 32);
        if (*(const int*)(smem + BESTE_OFF + arow * 4) != e) {   // mask rows not won by e
#pragma unroll
            for (int v = 0; v < 8; ++v) a.u[v] = 0u;
        }
#pragma unroll
        for (int j = 0; j < 6; ++j) {
            Frag16 b;
            const char* bp = smem + wp + ((j1 + j) * 16 + rr) * WP_STRIDE + hsel * 16;
            b.q[0] = *(const uint4*)bp;
            b.q[1] = *(const uint4*)(bp + 32);
            acc3[j] = __builtin_amdgcn_wmma_f32_16x16x32_bf16(
                false, a.v, false, b.v, (short)0, acc3[j], false, false);
        }
        __syncthreads();
    }

#pragma unroll
    for (int j = 0; j < 6; ++j) {
        int col = (j1 + j) * 16 + rr;
#pragma unroll
        for (int v = 0; v < 8; ++v) {
            int row = m1 * 16 + (hsel ? v + 8 : v);
            out[(size_t)(rb + row) * D_ + col] = acc3[j][v];
        }
    }
#undef LDSA
}

extern "C" void kernel_launch(void* const* d_in, const int* in_sizes, int n_in,
                              void* d_out, int out_size, void* d_ws, size_t ws_size,
                              hipStream_t stream) {
    (void)in_sizes; (void)n_in; (void)out_size; (void)ws_size;
    const float* x         = (const float*)d_in[0];
    const float* state_re  = (const float*)d_in[1];
    const float* state_im  = (const float*)d_in[2];
    const float* A_align   = (const float*)d_in[3];
    const float* W_in      = (const float*)d_in[4];
    const float* W_dt      = (const float*)d_in[5];
    const float* W_phase   = (const float*)d_in[6];
    const float* W_out     = (const float*)d_in[7];
    const float* log_omega = (const float*)d_in[8];
    const float* log_temp  = (const float*)d_in[9];
    const float* nu        = (const float*)d_in[10];
    float* out = (float*)d_out;
    unsigned char* ws = (unsigned char*)d_ws;

    // prep: convert weights/x to bf16 WMMA panel layouts in workspace
    rm3_prep_weights<<<dim3(3264), dim3(256), 0, stream>>>(A_align, W_in, W_dt, W_phase, W_out, ws);
    rm3_prep_x<<<dim3((B_ * D_ / 2) / 256), dim3(256), 0, stream>>>(x, ws);

    hipFuncSetAttribute(reinterpret_cast<const void*>(rm3_expert_pool_kernel),
                        hipFuncAttributeMaxDynamicSharedMemorySize, LDS_TOTAL);
    rm3_expert_pool_kernel<<<dim3(B_ / BM), dim3(256), LDS_TOTAL, stream>>>(
        ws, state_re, state_im, log_omega, log_temp, nu, out);
}
